// SwitchFeedForwardLoRALatent_9929964389241
// MI455X (gfx1250) — matrix operand order repaired and checked by reading, not compiled, not run on hardware
//
#include <hip/hip_runtime.h>
#include <hip/hip_bf16.h>
#include <math.h>

// ---------------------------------------------------------------------------
// SwitchFeedForwardLoRALatent for MI455X (gfx1250, wave32, WMMA bf16 path)
// Top-1 routed LoRA FFN: router GEMM + gather + grouped GEMMs, all via
// v_wmma_f32_16x16x32_bf16.  Memory-bound problem (~155MB @ 23.3TB/s ~ 7us).
// ---------------------------------------------------------------------------

typedef __attribute__((ext_vector_type(16))) __bf16 v16bf;
typedef __attribute__((ext_vector_type(16))) float  v16f;
typedef __attribute__((ext_vector_type(8)))  float  v8f;

#define H_DIM 768
#define R_DIM 64
#define E_NUM 8
#define T_TOK 16384          // B*S = 8*2048
#define KT_H  24             // 768/32 k-steps
#define NT_R  4              // 64/16 n-tiles
#define KT_R  2              // 64/32 k-steps
#define NT_H  48             // 768/16 n-tiles
#define TILE  16
#define MAXTILES (T_TOK/TILE + E_NUM)   // 1032 (padding worst case)
#define PERMCAP  (T_TOK + E_NUM*TILE)   // 16512

// bf16 elements per swizzled fragment matrix
#define ENC_FRAG_ELTS (KT_H*NT_R*32*16)   // 49152
#define W1_FRAG_ELTS  (KT_H*NT_R*32*16)   // per expert
#define W2_FRAG_ELTS  (KT_R*NT_H*32*16)   // per expert

// ----------------------------- helpers -------------------------------------

__device__ __forceinline__ unsigned short f2bfu(float f) {
  // round-to-nearest-even fp32 -> bf16 (scalar path, prep kernel only)
  unsigned u = __builtin_bit_cast(unsigned, f);
  u += 0x7FFFu + ((u >> 16) & 1u);
  return (unsigned short)(u >> 16);
}

union Frag16 { uint4 u[2]; v16bf v; };
union FragS  { unsigned short s[16]; v16bf v; };
union FragF  { float4 f4[4]; v16f v; };

// B-fragment: 16x32 k-tile x 16 n-tile, pre-swizzled in workspace.
// lane n=(l&15), hi=(l>>4): VGPRs 0-3 K = hi*8 + {0..7}; VGPRs 4-7 K = 16 + hi*8 + {0..7}
__device__ __forceinline__ v16bf load_bfrag(const uint4* base, int frag, int lane) {
  Frag16 f;
  const uint4* p = base + ((size_t)frag * 32 + lane) * 2;
  f.u[0] = p[0];
  f.u[1] = p[1];
  return f.v;
}

// A-fragment from a row-major fp32 row (K contiguous): lane row m=(l&15),
// hi=(l>>4): VGPRs 0-3 K = hi*8 + {0..7}; VGPRs 4-7 K = 16 + hi*8 + {0..7}
// Native packed converts: v_cvt_pk_bf16_f32 (RNE), 8 VALU per fragment.
__device__ __forceinline__ v16bf make_afrag(const float* base, int hi) {
  const float4* p0 = reinterpret_cast<const float4*>(base + hi * 8);
  const float4* p1 = reinterpret_cast<const float4*>(base + 16 + hi * 8);
  FragF f;
  f.f4[0] = p0[0];
  f.f4[1] = p0[1];
  f.f4[2] = p1[0];
  f.f4[3] = p1[1];
  return __builtin_convertvector(f.v, v16bf);
}

__device__ __forceinline__ v16bf zero_afrag() {
  FragS f;
#pragma unroll
  for (int i = 0; i < 16; ++i) f.s[i] = 0;
  return f.v;
}

#define WMMA_BF16(A, B, C) \
  __builtin_amdgcn_wmma_f32_16x16x32_bf16(false, (A), false, (B), (short)0, (C), false, false)

// ------------------------ kernel 1: weight prep -----------------------------
// Swizzle a row-major [K][N] fp32 matrix into bf16 B-fragments.
__device__ void prep_mat(const float* __restrict__ W, int K, int N,
                         unsigned short* __restrict__ dst, int tid, int stride) {
  int KT = K / 32, NT = N / 16;
  int total = KT * NT * 64;  // uint4 units: frags * 32 lanes * 2 halves
  for (int u = tid; u < total; u += stride) {
    int h2 = u & 1, lane = (u >> 1) & 31, f = u >> 6;
    int kt = f % KT, nt = f / KT;
    int n = nt * 16 + (lane & 15), hi = lane >> 4;
    int kbase = kt * 32 + h2 * 16 + hi * 8;
    unsigned short* d = dst + (size_t)u * 8;
#pragma unroll
    for (int j = 0; j < 8; ++j)
      d[j] = f2bfu(W[(size_t)(kbase + j) * N + n]);
  }
}

__global__ void prep_kernel(const float* __restrict__ enc_w,
                            const float* __restrict__ w1,
                            const float* __restrict__ w2,
                            unsigned short* __restrict__ enc_f,
                            unsigned short* __restrict__ w1_f,
                            unsigned short* __restrict__ w2_f,
                            int* __restrict__ meta, int* __restrict__ perm) {
  int tid = blockIdx.x * blockDim.x + threadIdx.x;
  int stride = gridDim.x * blockDim.x;
  prep_mat(enc_w, H_DIM, R_DIM, enc_f, tid, stride);
  for (int e = 0; e < E_NUM; ++e)
    prep_mat(w1 + (size_t)e * H_DIM * R_DIM, H_DIM, R_DIM,
             w1_f + (size_t)e * W1_FRAG_ELTS, tid, stride);
  for (int e = 0; e < E_NUM; ++e)
    prep_mat(w2 + (size_t)e * R_DIM * H_DIM, R_DIM, H_DIM,
             w2_f + (size_t)e * W2_FRAG_ELTS, tid, stride);
  for (int i = tid; i < PERMCAP; i += stride) perm[i] = -1;
  if (tid < 16) meta[tid] = 0;  // counts[0..7] + cursors[8..15]
}

// ------------------------ kernel 2: router ----------------------------------
__global__ __launch_bounds__(32) void router_kernel(
    const float* __restrict__ x, const uint4* __restrict__ enc_f,
    const float* __restrict__ enc_b, const float* __restrict__ sw_w,
    const float* __restrict__ sw_b, int* __restrict__ routes,
    float* __restrict__ probs, int* __restrict__ counts) {
  __shared__ float xe[TILE][R_DIM];
  int lane = threadIdx.x;
  int m = lane & 15, hi = lane >> 4;
  int t0 = blockIdx.x * TILE;
  const float* xrow = x + (size_t)(t0 + m) * H_DIM;

  v8f acc[NT_R] = {};
  for (int kt = 0; kt < KT_H; ++kt) {
    // issue all B-fragment loads first (one clause), overlap with cvt work
    v16bf b[NT_R];
#pragma unroll
    for (int nt = 0; nt < NT_R; ++nt)
      b[nt] = load_bfrag(enc_f, nt * KT_H + kt, lane);
    if (kt + 1 < KT_H)
      __builtin_prefetch(xrow + (kt + 1) * 32, 0, 0);
    v16bf a = make_afrag(xrow + kt * 32, hi);
#pragma unroll
    for (int nt = 0; nt < NT_R; ++nt)
      acc[nt] = WMMA_BF16(a, b[nt], acc[nt]);
  }
  int cn = lane & 15;
#pragma unroll
  for (int nt = 0; nt < NT_R; ++nt) {
    float bias = enc_b[nt * 16 + cn];
#pragma unroll
    for (int r = 0; r < 8; ++r)
      xe[hi * 8 + r][nt * 16 + cn] = acc[nt][r] + bias;
  }
  __syncthreads();

  if (lane < TILE) {
    float lg[E_NUM];
#pragma unroll
    for (int e = 0; e < E_NUM; ++e) lg[e] = sw_b[e];
    for (int r = 0; r < R_DIM; ++r) {
      float xv = xe[lane][r];
#pragma unroll
      for (int e = 0; e < E_NUM; ++e) lg[e] += xv * sw_w[r * E_NUM + e];
    }
    float mx = lg[0]; int arg = 0;
#pragma unroll
    for (int e = 1; e < E_NUM; ++e)
      if (lg[e] > mx) { mx = lg[e]; arg = e; }
    float s = 0.f;
#pragma unroll
    for (int e = 0; e < E_NUM; ++e) s += expf(lg[e] - mx);
    int t = t0 + lane;
    routes[t] = arg;
    probs[t] = 1.0f / s;       // max softmax prob
    atomicAdd(&counts[arg], 1);
  }
}

// ------------------- kernel 3: padded segment offsets -----------------------
__global__ void offsets_kernel(int* __restrict__ meta) {
  if (threadIdx.x == 0 && blockIdx.x == 0) {
    int off = 0;
    for (int e = 0; e < E_NUM; ++e) {
      meta[16 + e] = off;
      off += ((meta[e] + TILE - 1) / TILE) * TILE;
    }
    meta[16 + E_NUM] = off;  // padded total
  }
}

// ------------------------ kernel 4: scatter ---------------------------------
__global__ void scatter_kernel(const int* __restrict__ routes,
                               int* __restrict__ meta, int* __restrict__ perm) {
  int t = blockIdx.x * blockDim.x + threadIdx.x;
  if (t >= T_TOK) return;
  int e = routes[t];
  int r = atomicAdd(&meta[8 + e], 1);
  perm[meta[16 + e] + r] = t;
}

// -------------- kernel 5: h = gelu(x @ w1[e] + b1[e]) -----------------------
__global__ __launch_bounds__(32) void ffn1_kernel(
    const float* __restrict__ x, const uint4* __restrict__ w1_f,
    const float* __restrict__ b1, const int* __restrict__ meta,
    const int* __restrict__ perm, float* __restrict__ hbuf) {
  int g0 = blockIdx.x * TILE;
  if (g0 >= meta[16 + E_NUM]) return;
  int e = 0;
  while (e < E_NUM - 1 && g0 >= meta[16 + e + 1]) ++e;

  int lane = threadIdx.x, m = lane & 15, hi = lane >> 4;
  int tok = perm[g0 + m];
  const float* xrow = x + (size_t)(tok >= 0 ? tok : 0) * H_DIM;
  bool valid = (tok >= 0);
  const uint4* wf = w1_f + (size_t)e * (W1_FRAG_ELTS / 8);

  v8f acc[NT_R] = {};
  for (int kt = 0; kt < KT_H; ++kt) {
    v16bf b[NT_R];
#pragma unroll
    for (int nt = 0; nt < NT_R; ++nt)
      b[nt] = load_bfrag(wf, nt * KT_H + kt, lane);
    if (valid && kt + 1 < KT_H)
      __builtin_prefetch(xrow + (kt + 1) * 32, 0, 0);
    v16bf a;
    if (valid) a = make_afrag(xrow + kt * 32, hi);
    else       a = zero_afrag();
#pragma unroll
    for (int nt = 0; nt < NT_R; ++nt)
      acc[nt] = WMMA_BF16(a, b[nt], acc[nt]);
  }
  int cn = lane & 15;
#pragma unroll
  for (int nt = 0; nt < NT_R; ++nt) {
    float bias = b1[e * R_DIM + nt * 16 + cn];
#pragma unroll
    for (int r = 0; r < 8; ++r) {
      float v = acc[nt][r] + bias;
      v = 0.5f * v * (1.0f + erff(v * 0.70710678118654752f));  // exact gelu
      hbuf[(size_t)(g0 + hi * 8 + r) * R_DIM + nt * 16 + cn] = v;
    }
  }
}

// ------ kernel 6: out = (h @ w2[e] + b2[e]) * p, scattered to tokens --------
__global__ __launch_bounds__(256) void ffn2_kernel(
    const float* __restrict__ hbuf, const uint4* __restrict__ w2_f,
    const float* __restrict__ b2, const float* __restrict__ probs,
    const int* __restrict__ meta, const int* __restrict__ perm,
    float* __restrict__ out) {
  int g0 = blockIdx.x * TILE;
  if (g0 >= meta[16 + E_NUM]) return;
  int e = 0;
  while (e < E_NUM - 1 && g0 >= meta[16 + e + 1]) ++e;

  int wave = threadIdx.x >> 5, lane = threadIdx.x & 31;
  int m = lane & 15, hi = lane >> 4;
  const float* hrow = hbuf + (size_t)(g0 + m) * R_DIM;
  v16bf a0 = make_afrag(hrow + 0, hi);
  v16bf a1 = make_afrag(hrow + 32, hi);
  const uint4* wf = w2_f + (size_t)e * (W2_FRAG_ELTS / 8);

  int tok[8]; float pr[8];
#pragma unroll
  for (int r = 0; r < 8; ++r) {
    tok[r] = perm[g0 + hi * 8 + r];
    pr[r] = (tok[r] >= 0) ? probs[tok[r]] : 0.f;
  }

  int cn = lane & 15;
#pragma unroll
  for (int j = 0; j < 6; ++j) {
    int nt = wave * 6 + j;               // 8 waves x 6 = 48 n-tiles of 768
    v16bf b0 = load_bfrag(wf, nt * KT_R + 0, lane);
    v16bf b1v = load_bfrag(wf, nt * KT_R + 1, lane);
    v8f acc = {};
    acc = WMMA_BF16(a0, b0, acc);
    acc = WMMA_BF16(a1, b1v, acc);
    int col = nt * 16 + cn;
    float bias = b2[e * H_DIM + col];
#pragma unroll
    for (int r = 0; r < 8; ++r) {
      if (tok[r] >= 0)
        out[(size_t)tok[r] * H_DIM + col] = (acc[r] + bias) * pr[r];
    }
  }
}

// ----------------------------- launcher -------------------------------------
extern "C" void kernel_launch(void* const* d_in, const int* in_sizes, int n_in,
                              void* d_out, int out_size, void* d_ws, size_t ws_size,
                              hipStream_t stream) {
  const float* x     = (const float*)d_in[0];
  const float* enc_w = (const float*)d_in[1];
  const float* enc_b = (const float*)d_in[2];
  const float* sw_w  = (const float*)d_in[3];
  const float* sw_b  = (const float*)d_in[4];
  const float* w1    = (const float*)d_in[5];
  const float* b1    = (const float*)d_in[6];
  const float* w2    = (const float*)d_in[7];
  const float* b2    = (const float*)d_in[8];
  float* out = (float*)d_out;

  // workspace layout (all offsets multiples of 256B); total ~6.0 MB
  char* ws = (char*)d_ws;
  size_t o_enc = 0;
  size_t o_w1  = o_enc + (size_t)ENC_FRAG_ELTS * 2;
  size_t o_w2  = o_w1 + (size_t)E_NUM * W1_FRAG_ELTS * 2;
  size_t o_rt  = o_w2 + (size_t)E_NUM * W2_FRAG_ELTS * 2;
  size_t o_pb  = o_rt + (size_t)T_TOK * 4;
  size_t o_mt  = o_pb + (size_t)T_TOK * 4;
  size_t o_pm  = o_mt + 256;
  size_t o_h   = o_pm + (size_t)PERMCAP * 4;

  unsigned short* enc_f = (unsigned short*)(ws + o_enc);
  unsigned short* w1_f  = (unsigned short*)(ws + o_w1);
  unsigned short* w2_f  = (unsigned short*)(ws + o_w2);
  int*   routes = (int*)(ws + o_rt);
  float* probs  = (float*)(ws + o_pb);
  int*   meta   = (int*)(ws + o_mt);   // [0..7]=counts [8..15]=cursors [16..24]=offsets
  int*   perm   = (int*)(ws + o_pm);
  float* hbuf   = (float*)(ws + o_h);

  prep_kernel<<<256, 256, 0, stream>>>(enc_w, w1, w2, enc_f, w1_f, w2_f, meta, perm);
  router_kernel<<<T_TOK / TILE, 32, 0, stream>>>(
      x, (const uint4*)enc_f, enc_b, sw_w, sw_b, routes, probs, meta);
  offsets_kernel<<<1, 32, 0, stream>>>(meta);
  scatter_kernel<<<(T_TOK + 255) / 256, 256, 0, stream>>>(routes, meta, perm);
  ffn1_kernel<<<MAXTILES, 32, 0, stream>>>(x, (const uint4*)w1_f, b1, meta, perm, hbuf);
  ffn2_kernel<<<MAXTILES, 256, 0, stream>>>(hbuf, (const uint4*)w2_f, b2, probs, meta, perm, out);
}
